// CausalSelfAttention_53781580480569
// MI455X (gfx1250) — matrix operand order repaired and checked by reading, not compiled, not run on hardware
//
#include <hip/hip_runtime.h>

// ---------------------------------------------------------------------------
// CDNA5 (gfx1250) causal GQA self-attention layer.
// bf16 WMMA (v_wmma_f32_16x16x32_bf16) + async global->LDS staging (ASYNCcnt).
// ---------------------------------------------------------------------------

typedef __attribute__((ext_vector_type(16))) __bf16 v16bf;
typedef __attribute__((ext_vector_type(8)))  __bf16 v8bf;
typedef __attribute__((ext_vector_type(8)))  float  v8f;

#define HIDDEN 2048
#define NH     16
#define NKV    4
#define HD     128
#define BATCH  2
#define SEQ    2048
#define MTOK   (BATCH * SEQ)          // 4096 tokens

// ---------------------------------------------------------------------------
// WMMA helpers
// ---------------------------------------------------------------------------
__device__ __forceinline__ v8f wmma_bf16(const v16bf& a, const v16bf& b, const v8f& c) {
  // D = A(16x32) * B(32x16) + C, f32 accumulate
  return __builtin_amdgcn_wmma_f32_16x16x32_bf16(false, a, false, b, (short)0, c, false, false);
}

__device__ __forceinline__ v16bf cat16(v8bf lo, v8bf hi) {
  v16bf r;
#pragma unroll
  for (int i = 0; i < 8; ++i) { r[i] = lo[i]; r[i + 8] = hi[i]; }
  return r;
}

// A-fragment (16x32 bf16, MxK), source row-major with row stride ld (elems).
// Lane L(0-15): row=L, K=0..7 & 16..23 ; Lane L(16-31): row=L-16, K=8..15 & 24..31
__device__ __forceinline__ v16bf load_a_frag(const __bf16* base, int ld, int lane) {
  const __bf16* p = base + (lane & 15) * ld + ((lane & 16) ? 8 : 0);
  return cat16(*(const v8bf*)p, *(const v8bf*)(p + 16));
}

// B-fragment (32x16 bf16, KxN) from a K-transposed tile Bt[n][k] (stride ld elems).
// Lane L(0-15): col=L, K=0..7 & 16..23 ; Lane L(16-31): col=L-16, K=8..15 & 24..31
__device__ __forceinline__ v16bf load_bt_frag(const __bf16* base, int ld, int lane) {
  const __bf16* p = base + (lane & 15) * ld + ((lane & 16) ? 8 : 0);
  return cat16(*(const v8bf*)p, *(const v8bf*)(p + 16));
}

// ---------------------------------------------------------------------------
// Async global->LDS helpers (gfx1250 GLOBAL_LOAD_ASYNC_TO_LDS_B128, ASYNCcnt)
// ---------------------------------------------------------------------------
__device__ __forceinline__ unsigned lds_off(const void* p) {
  // generic -> LDS(AS3) addrspacecast yields the 0-based LDS byte address
  return (unsigned)(unsigned long long)(__attribute__((address_space(3))) const void*)p;
}

__device__ __forceinline__ void async_load_b128(unsigned lds_byte_addr,
                                                unsigned glb_byte_off,
                                                unsigned long long base) {
  // GVS form: mem = SGPR_base + VGPR_u32_offset ; dst LDS addr from VGPR
  asm volatile("global_load_async_to_lds_b128 %0, %1, %2"
               :: "v"(lds_byte_addr), "v"(glb_byte_off), "s"(base)
               : "memory");
}

__device__ __forceinline__ void wait_async0() {
  asm volatile("s_wait_asynccnt 0" ::: "memory");
}

// ---------------------------------------------------------------------------
// f32 -> bf16 conversion
// ---------------------------------------------------------------------------
__global__ __launch_bounds__(256) void cvt_f32_bf16(const float* __restrict__ src,
                                                    __bf16* __restrict__ dst, int n) {
  for (int i = blockIdx.x * blockDim.x + threadIdx.x; i < n; i += gridDim.x * blockDim.x)
    dst[i] = (__bf16)src[i];
}

// ---------------------------------------------------------------------------
// Tiled GEMM: C[MxN] = A[MxK](bf16) * B[KxN](bf16) + bias
// BM=128, BN=128, BK=64. 8 waves: 4(M) x 2(N); each wave 32x64 output.
// A tile staged with async global->LDS; B tile staged transposed (sync scatter).
// ---------------------------------------------------------------------------
template <bool OUTF32>
__global__ __launch_bounds__(256) void gemm_bf16(const __bf16* __restrict__ A,
                                                 const __bf16* __restrict__ B,
                                                 const float* __restrict__ bias,
                                                 void* __restrict__ Cout,
                                                 int M, int N, int K) {
  __shared__ __bf16 As[128][80];   // A tile, row-major, pad->160B row stride
  __shared__ __bf16 Bs[128][80];   // B tile transposed: Bs[n][k]

  const int tid  = threadIdx.x;
  const int lane = tid & 31;
  const int wave = tid >> 5;
  const int wm   = wave & 3;       // M quadrant: rows wm*32
  const int wn   = wave >> 2;      // N half:     cols wn*64
  const int m0   = blockIdx.y * 128;
  const int n0   = blockIdx.x * 128;

  const unsigned long long Abase = (unsigned long long)A;

  v8f acc[2][4];
#pragma unroll
  for (int i = 0; i < 2; ++i)
#pragma unroll
    for (int j = 0; j < 4; ++j) acc[i][j] = {};

  for (int k0 = 0; k0 < K; k0 += 64) {
    __syncthreads();
    // A tile: 128x64 = 1024 x 16B chunks, async global->LDS
#pragma unroll
    for (int c = tid; c < 1024; c += 256) {
      int r = c >> 3, col = (c & 7) * 8;
      unsigned goff = (unsigned)(((m0 + r) * K + k0 + col) * 2);
      async_load_b128(lds_off(&As[r][col]), goff, Abase);
    }
    // B tile: 64x128, store transposed (sync)
#pragma unroll
    for (int c = tid; c < 1024; c += 256) {
      int kk = c >> 4, n = (c & 15) * 8;
      v8bf x = *(const v8bf*)(B + (size_t)(k0 + kk) * N + n0 + n);
#pragma unroll
      for (int i = 0; i < 8; ++i) Bs[n + i][kk] = x[i];
    }
    wait_async0();
    __syncthreads();

#pragma unroll
    for (int ks = 0; ks < 2; ++ks) {
      v16bf a0 = load_a_frag(&As[wm * 32][ks * 32], 80, lane);
      v16bf a1 = load_a_frag(&As[wm * 32 + 16][ks * 32], 80, lane);
#pragma unroll
      for (int fn = 0; fn < 4; ++fn) {
        v16bf bf = load_bt_frag(&Bs[wn * 64 + fn * 16][ks * 32], 80, lane);
        acc[0][fn] = wmma_bf16(a0, bf, acc[0][fn]);
        acc[1][fn] = wmma_bf16(a1, bf, acc[1][fn]);
      }
    }
  }

  // Epilogue: C layout — vgpr m: lanes 0-15 row m, lanes 16-31 row m+8; col = lane&15
  const int nl = lane & 15;
  const int hi8 = (lane & 16) ? 8 : 0;
#pragma unroll
  for (int fm = 0; fm < 2; ++fm)
#pragma unroll
    for (int fn = 0; fn < 4; ++fn)
#pragma unroll
      for (int m = 0; m < 8; ++m) {
        int gr = m0 + wm * 32 + fm * 16 + hi8 + m;
        int gc = n0 + wn * 64 + fn * 16 + nl;
        float v = acc[fm][fn][m] + bias[gc];
        if (OUTF32) ((float*)Cout)[(size_t)gr * N + gc] = v;
        else        ((__bf16*)Cout)[(size_t)gr * N + gc] = (__bf16)v;
      }
}

// ---------------------------------------------------------------------------
// Flash attention, causal + key mask, GQA (head h uses kv head h/4).
// Q: [B, S, NH*HD] bf16 (row stride 2048), K/V: [B, S, NKV*HD] bf16 (stride 512)
// Block: 128 q-rows, 8 waves x 16 rows. Key tiles of 64.
// K tile staged async (natural layout = B^T for Q*K^T); V staged transposed.
// ---------------------------------------------------------------------------
__global__ __launch_bounds__(256) void attn_kernel(const __bf16* __restrict__ Q,
                                                   const __bf16* __restrict__ Kg,
                                                   const __bf16* __restrict__ Vg,
                                                   const int* __restrict__ amask,
                                                   __bf16* __restrict__ O) {
  __shared__ __bf16 Ks[64][144];      // K tile [key][d]   (natural = B^T for Q*K^T)
  __shared__ __bf16 Vt[128][80];      // V tile transposed [d][key]
  __shared__ __bf16 Ps[8][16][80];    // per-wave P (16 x 64), C-layout -> A-layout relay

  const int tid  = threadIdx.x;
  const int lane = tid & 31;
  const int wave = tid >> 5;
  const int q0   = blockIdx.x * 128;
  const int bh   = blockIdx.y;
  const int b    = bh >> 4;
  const int h    = bh & 15;
  const int kvh  = h >> 2;
  const int nl   = lane & 15;
  const int hi8  = (lane & 16) ? 8 : 0;
  const int qrow0 = q0 + wave * 16;

  const unsigned long long Kbase = (unsigned long long)Kg;

  // Q fragments straight from global (16 rows x 128 d = 4 A-frags)
  v16bf qa[4];
#pragma unroll
  for (int kk = 0; kk < 4; ++kk) {
    const __bf16* p = Q + ((size_t)(b * SEQ + qrow0 + nl)) * (NH * HD)
                        + h * HD + kk * 32 + ((lane & 16) ? 8 : 0);
    qa[kk] = cat16(*(const v8bf*)p, *(const v8bf*)(p + 16));
  }

  v8f oacc[8];
#pragma unroll
  for (int i = 0; i < 8; ++i) oacc[i] = {};
  float mrow[8], lrow[8];
#pragma unroll
  for (int i = 0; i < 8; ++i) { mrow[i] = -1e30f; lrow[i] = 0.0f; }

  const float scale = 0.08838834764831843f;  // 1/sqrt(128)
  const int ntiles = q0 / 64 + 2;            // causal: keys up to q0+127

  for (int t = 0; t < ntiles; ++t) {
    const int kt0 = t * 64;
    __syncthreads();
    // stage K tile (async, natural) and V tile (sync, transposed)
#pragma unroll
    for (int c = tid; c < 1024; c += 256) {
      int ky = c >> 4, d = (c & 15) * 8;
      unsigned ge = (unsigned)(((b * SEQ + kt0 + ky) * (NKV * HD) + kvh * HD + d));
      async_load_b128(lds_off(&Ks[ky][d]), ge * 2u, Kbase);
      v8bf xv = *(const v8bf*)(Vg + (size_t)ge);
#pragma unroll
      for (int i = 0; i < 8; ++i) Vt[d + i][ky] = xv[i];
    }
    wait_async0();
    __syncthreads();

    // S = Q * K^T : 4 K-steps(d) x 4 N-frags(keys) = 16 wmma
    v8f sacc[4];
#pragma unroll
    for (int i = 0; i < 4; ++i) sacc[i] = {};
#pragma unroll
    for (int kk = 0; kk < 4; ++kk)
#pragma unroll
      for (int fn = 0; fn < 4; ++fn) {
        v16bf bk = load_bt_frag(&Ks[fn * 16][kk * 32], 144, lane);
        sacc[fn] = wmma_bf16(qa[kk], bk, sacc[fn]);
      }

    // scale + causal/key mask (value selects; EXEC stays full)
#pragma unroll
    for (int fn = 0; fn < 4; ++fn) {
      int kr = kt0 + fn * 16 + nl;
      bool okm = (amask[b * SEQ + kr] != 0);
#pragma unroll
      for (int m = 0; m < 8; ++m) {
        int qr = qrow0 + hi8 + m;
        float sv = sacc[fn][m] * scale;
        sacc[fn][m] = (okm && kr <= qr) ? sv : -1e30f;
      }
    }

    // online softmax (16-lane halves reduce independently; xor<=8 stays in half)
    __bf16* Pw = &Ps[wave][0][0];
#pragma unroll
    for (int m = 0; m < 8; ++m) {
      float rmax = sacc[0][m];
#pragma unroll
      for (int fn = 1; fn < 4; ++fn) rmax = fmaxf(rmax, sacc[fn][m]);
#pragma unroll
      for (int off = 1; off < 16; off <<= 1)
        rmax = fmaxf(rmax, __shfl_xor(rmax, off, 32));
      float mnew = fmaxf(mrow[m], rmax);
      float corr = __expf(mrow[m] - mnew);
      float ps = 0.0f;
#pragma unroll
      for (int fn = 0; fn < 4; ++fn) {
        float p = __expf(sacc[fn][m] - mnew);
        ps += p;
        Pw[(hi8 + m) * 80 + fn * 16 + nl] = (__bf16)p;   // C-layout -> LDS
      }
#pragma unroll
      for (int off = 1; off < 16; off <<= 1)
        ps += __shfl_xor(ps, off, 32);
      lrow[m] = lrow[m] * corr + ps;
      mrow[m] = mnew;
#pragma unroll
      for (int fo = 0; fo < 8; ++fo) oacc[fo][m] *= corr;
    }

    // same-wave LDS RAW: wait our ds stores before re-reading P as A-frags
    asm volatile("s_wait_dscnt 0" ::: "memory");

    // O += P * V : 2 K-steps(keys) x 8 N-frags(d) = 16 wmma
#pragma unroll
    for (int kc = 0; kc < 2; ++kc) {
      v16bf pa = load_a_frag(Pw + kc * 32, 80, lane);
#pragma unroll
      for (int fo = 0; fo < 8; ++fo) {
        v16bf vb = load_bt_frag(&Vt[fo * 16][kc * 32], 80, lane);
        oacc[fo] = wmma_bf16(pa, vb, oacc[fo]);
      }
    }
  }

  // normalize + store attn output [B, S, NH*HD] bf16
#pragma unroll
  for (int fo = 0; fo < 8; ++fo)
#pragma unroll
    for (int m = 0; m < 8; ++m) {
      int qr = qrow0 + hi8 + m;
      float ov = oacc[fo][m] / lrow[m];
      O[((size_t)(b * SEQ + qr)) * (NH * HD) + h * HD + fo * 16 + nl] = (__bf16)ov;
    }
}

// ---------------------------------------------------------------------------
// Launch
// ---------------------------------------------------------------------------
extern "C" void kernel_launch(void* const* d_in, const int* in_sizes, int n_in,
                              void* d_out, int out_size, void* d_ws, size_t ws_size,
                              hipStream_t stream) {
  const float* X     = (const float*)d_in[0];
  const int*   amask = (const int*)d_in[1];
  const float* Wq = (const float*)d_in[2];
  const float* bq = (const float*)d_in[3];
  const float* Wk = (const float*)d_in[4];
  const float* bk = (const float*)d_in[5];
  const float* Wv = (const float*)d_in[6];
  const float* bv = (const float*)d_in[7];
  const float* Wo = (const float*)d_in[8];
  const float* bo = (const float*)d_in[9];

  char* ws = (char*)d_ws;
  size_t off = 0;
  auto alloc = [&](size_t bytes) -> void* {
    void* p = ws + off;
    off = (off + bytes + 255) & ~(size_t)255;
    return p;
  };

  __bf16* Xb  = (__bf16*)alloc((size_t)MTOK * HIDDEN * 2);
  __bf16* Wqb = (__bf16*)alloc((size_t)HIDDEN * 2048 * 2);
  __bf16* Wkb = (__bf16*)alloc((size_t)HIDDEN * 512 * 2);
  __bf16* Wvb = (__bf16*)alloc((size_t)HIDDEN * 512 * 2);
  __bf16* Wob = (__bf16*)alloc((size_t)2048 * HIDDEN * 2);
  __bf16* Qb  = (__bf16*)alloc((size_t)MTOK * 2048 * 2);
  __bf16* Kb  = (__bf16*)alloc((size_t)MTOK * 512 * 2);
  __bf16* Vb  = (__bf16*)alloc((size_t)MTOK * 512 * 2);
  __bf16* Ab  = (__bf16*)alloc((size_t)MTOK * 2048 * 2);

  cvt_f32_bf16<<<1024, 256, 0, stream>>>(X,  Xb,  MTOK * HIDDEN);
  cvt_f32_bf16<<<1024, 256, 0, stream>>>(Wq, Wqb, HIDDEN * 2048);
  cvt_f32_bf16<<<1024, 256, 0, stream>>>(Wk, Wkb, HIDDEN * 512);
  cvt_f32_bf16<<<1024, 256, 0, stream>>>(Wv, Wvb, HIDDEN * 512);
  cvt_f32_bf16<<<1024, 256, 0, stream>>>(Wo, Wob, 2048 * HIDDEN);

  // QKV projections (bf16 out, bias fused)
  gemm_bf16<false><<<dim3(2048 / 128, MTOK / 128), 256, 0, stream>>>(Xb, Wqb, bq, Qb, MTOK, 2048, HIDDEN);
  gemm_bf16<false><<<dim3(512  / 128, MTOK / 128), 256, 0, stream>>>(Xb, Wkb, bk, Kb, MTOK, 512,  HIDDEN);
  gemm_bf16<false><<<dim3(512  / 128, MTOK / 128), 256, 0, stream>>>(Xb, Wvb, bv, Vb, MTOK, 512,  HIDDEN);

  // flash attention
  attn_kernel<<<dim3(SEQ / 128, BATCH * NH), 256, 0, stream>>>(Qb, Kb, Vb, amask, Ab);

  // output projection (f32 out)
  gemm_bf16<true><<<dim3(2048 / 128, MTOK / 128), 256, 0, stream>>>(Ab, Wob, bo, d_out, MTOK, 2048, HIDDEN);
}
